// xLSTM_86844238725689
// MI455X (gfx1250) — matrix operand order repaired and checked by reading, not compiled, not run on hardware
//
#include <hip/hip_runtime.h>
#include <hip/hip_bf16.h>

#define HDIM  128
#define G4H   512
#define TLEN  1024
#define BATCH 128
#define BT    16

#define VCHUNK 4                 // k-tiles per TDM chunk (4 KB)
#define VNCH   (512 / VCHUNK)    // 128 chunks per step per wave
#define VBYTES (VCHUNK * 32 * 16 * 2)

typedef __attribute__((ext_vector_type(16))) __bf16 bf16x16;
typedef __attribute__((ext_vector_type(8)))  __bf16 bf16x8;
typedef __attribute__((ext_vector_type(8)))  float  f32x8;
typedef __attribute__((ext_vector_type(4)))  unsigned int u32x4;
typedef __attribute__((ext_vector_type(8)))  int i32x8;
typedef __attribute__((ext_vector_type(4)))  int i32x4;

__device__ __forceinline__ f32x8 wmma_bf16(bf16x16 a, bf16x16 b, f32x8 c) {
  return __builtin_amdgcn_wmma_f32_16x16x32_bf16(false, a, false, b, (short)0, c,
                                                 false, false);
}

// A fragment (16x32 bf16, ISA 7.12.2): lane m (lo half) holds K {0..7,16..23},
// lane m+16 holds K {8..15,24..31}. Row pointer is k-contiguous bf16.
__device__ __forceinline__ bf16x16 load_afrag_bf(const __bf16* row, int kt, int hi) {
  const int k0 = kt * 32 + hi * 8;
  bf16x8 lo = *(const bf16x8*)(row + k0);
  bf16x8 hh = *(const bf16x8*)(row + k0 + 16);
  bf16x16 r;
#pragma unroll
  for (int j = 0; j < 8; ++j) { r[j] = lo[j]; r[8 + j] = hh[j]; }
  return r;
}

// Same A fragment but built from an fp32 row (converted to bf16 on the fly).
__device__ __forceinline__ bf16x16 make_afrag_f32(const float* row, int kt, int hi) {
  const int k0 = kt * 32 + hi * 8;
  bf16x16 r;
#pragma unroll
  for (int j = 0; j < 8; ++j) {
    r[j]     = (__bf16)row[k0 + j];
    r[8 + j] = (__bf16)row[k0 + 16 + j];
  }
  return r;
}

// B fragment (32x16 bf16): lane n<16 holds column n, K {0..7,16..23}; lane n+16 holds
// K {8..15,24..31}. Pre-packed so each lane reads 32 contiguous bytes. Layout [nt][kt][lane][16].
__device__ __forceinline__ bf16x16 load_bfrag(const __bf16* __restrict__ pack,
                                              int KT, int kt, int nt, int lane) {
  return *(const bf16x16*)(pack + ((((size_t)nt * KT + kt) * 32 + lane) << 4));
}

__device__ __forceinline__ float sigm(float x) {
  return 1.0f / (1.0f + __expf(-x));
}

// Flat LDS aperture: addr[31:0] is the LDS byte offset (ISA 10.2 aperture calc).
__device__ __forceinline__ unsigned lds_off(const void* p) {
  return (unsigned)(size_t)p;
}

// Tensor Data Mover: 1-D contiguous copy global -> LDS.
// D# per ISA 8.3-8.6: group0 = {count=1 | lds_addr | global_addr | type=2},
// group1 = {data_size=8B, tensor_dim0=n, tensor_dim1=1, tile_dim0=n, stride0=n}.
// This toolchain's builtin takes 6 operands (probe-confirmed arity); the 5th
// (extra int32x8 group) is passed zero-filled.
__device__ __forceinline__ void tdm_load_1d(unsigned lds_addr, const void* gsrc,
                                            unsigned nbytes) {
  const unsigned n = nbytes >> 3;              // elements of 8 bytes
  const unsigned long long ga = (unsigned long long)(size_t)gsrc;
  u32x4 g0;
  g0[0] = 1u;                                  // count=1, user descriptor
  g0[1] = lds_addr;                            // lds_addr (bytes)
  g0[2] = (unsigned)(ga & 0xffffffffu);        // global_addr[31:0]
  g0[3] = (unsigned)((ga >> 32) & 0x01ffffffu) // global_addr[56:32]
        | (2u << 30);                          // type = 2 ("image")
  i32x8 g1;
  g1[0] = (int)(3u << 16);                     // wg_mask=0, data_size=3 (8B)
  g1[1] = (int)((n & 0xffffu) << 16);          // tensor_dim0[15:0]
  g1[2] = (int)(((n >> 16) & 0xffffu) | (1u << 16)); // dim0[31:16], tensor_dim1=1
  g1[3] = (int)((n & 0xffffu) << 16);          // tile_dim0 = n
  g1[4] = 0;                                   // tile_dim1 = tile_dim2 = 0 (unused)
  g1[5] = (int)n;                              // tensor_dim0_stride[31:0]
  g1[6] = 0;
  g1[7] = 0;
  i32x4 g2 = {1, 0, 0, 0};                     // tensor_dim2 = 1
  i32x4 g3 = {0, 0, 0, 0};
  i32x8 g4 = {0, 0, 0, 0, 0, 0, 0, 0};         // unused extra group (zero)
  __builtin_amdgcn_tensor_load_to_lds(g0, g1, g2, g3, g4, 0);
}

// ---------------- packing / conversion kernels ----------------

__global__ void cvt_bf16_kernel(const float* __restrict__ in,
                                __bf16* __restrict__ out, size_t n) {
  size_t i = (size_t)blockIdx.x * blockDim.x + threadIdx.x;
  if (i < n) out[i] = (__bf16)in[i];
}

// Pack row-major fp32 B[K,N] into bf16 WMMA B-fragment order.
__global__ void pack_b_kernel(const float* __restrict__ B, __bf16* __restrict__ out,
                              int K, int N) {
  int idx = blockIdx.x * blockDim.x + threadIdx.x;
  const int KT = K >> 5;
  const int total = KT * (N >> 4) * 32;
  if (idx >= total) return;
  const int lane = idx & 31;
  const int frag = idx >> 5;
  const int kt = frag % KT;
  const int nt = frag / KT;
  const int n  = (nt << 4) + (lane & 15);
  const int kb = (kt << 5) + ((lane & 16) ? 8 : 0);
  __bf16* dst = out + ((size_t)idx << 4);
#pragma unroll
  for (int j = 0; j < 16; ++j)
    dst[j] = (__bf16)B[(size_t)(kb + j + (j & 8)) * N + n];
}

// ---------------- sLSTM persistent scan ----------------

__launch_bounds__(256, 1)
__global__ void slstm_kernel(const __bf16* __restrict__ x_bf,
                             const __bf16* __restrict__ Wpack,
                             const __bf16* __restrict__ Upack,
                             const float* __restrict__ Wb,
                             const float* __restrict__ Ub,
                             const float* __restrict__ alpha,
                             __bf16* __restrict__ hs_bf) {
  __shared__ __align__(32) float  gates[BT * G4H];   // 32 KB
  __shared__ __align__(32) float  c_s[BT * HDIM];    // 8 KB
  __shared__ __align__(32) __bf16 h_bf[BT * HDIM];   // 4 KB

  const int tid  = threadIdx.x;
  const int lane = tid & 31;
  const int wave = tid >> 5;
  const int m    = lane & 15;
  const int hi   = (lane >> 4) & 1;
  const int b0   = blockIdx.x * BT;

  for (int e = tid; e < BT * HDIM; e += 256) { c_s[e] = 0.f; h_bf[e] = (__bf16)0.f; }
  __syncthreads();

  const int hcol = tid & (HDIM - 1);
  const float bi = Wb[hcol]            + Ub[hcol];
  const float bf = Wb[HDIM + hcol]     + Ub[HDIM + hcol];
  const float bo = Wb[2 * HDIM + hcol] + Ub[2 * HDIM + hcol];
  const float bg = Wb[3 * HDIM + hcol] + Ub[3 * HDIM + hcol];
  const float al = alpha[hcol];

  for (int t = 0; t < TLEN; ++t) {
    const __bf16* xrow = x_bf + ((size_t)(b0 + m) * TLEN + t) * HDIM;
    if (t + 1 < TLEN) __builtin_prefetch(xrow + HDIM, 0, 1);  // global_prefetch_b8
    bf16x16 xf[4], hf[4];
#pragma unroll
    for (int kt = 0; kt < 4; ++kt) {
      xf[kt] = load_afrag_bf(xrow, kt, hi);
      hf[kt] = load_afrag_bf(&h_bf[m * HDIM], kt, hi);
    }
#pragma unroll
    for (int nti = 0; nti < 4; ++nti) {
      const int nt = wave * 4 + nti;
      f32x8 acc = {};
#pragma unroll
      for (int kt = 0; kt < 4; ++kt) {
        acc = wmma_bf16(xf[kt], load_bfrag(Wpack, 4, kt, nt, lane), acc);
        acc = wmma_bf16(hf[kt], load_bfrag(Upack, 4, kt, nt, lane), acc);
      }
      const int n = (nt << 4) + m;
#pragma unroll
      for (int r = 0; r < 8; ++r)            // C layout: VGPR r -> M = r + 8*hi
        gates[(r + hi * 8) * G4H + n] = acc[r];
    }
    __syncthreads();

    for (int e = tid; e < BT * HDIM; e += 256) {
      const int bb = e >> 7;
      const float* g = &gates[bb * G4H];
      const float iv = sigm(g[hcol] + bi);
      const float fv = sigm(g[HDIM + hcol] + bf);
      const float ov = sigm(g[2 * HDIM + hcol] + bo);
      const float gv = tanhf(g[3 * HDIM + hcol] + bg);
      const float cv = al * (fv * c_s[e] + iv * gv);
      c_s[e] = cv;
      const float hv = ov * tanhf(cv);
      h_bf[e] = (__bf16)hv;
      hs_bf[((size_t)(b0 + bb) * TLEN + t) * HDIM + hcol] = (__bf16)hv;
    }
    __syncthreads();
  }
}

// ---------------- mLSTM persistent scan ----------------
// Gate GEMM as in sLSTM, plus the dominant bilinear GEMM per step:
//   m = tanh( (x_t (x) h) @ Vw + Vb ),  K = 16384 -> 512 k-tiles.
// Vw (4 MB bf16, L2-resident) is streamed through per-wave LDS double buffers
// via the Tensor Data Mover, depth-1 pipelined with s_wait_tensorcnt. A
// fragments are built in VALU (one x scalar times a contiguous h slice),
// co-executing with the XDL WMMA pipe.

__launch_bounds__(256, 1)
__global__ void mlstm_kernel(const __bf16* __restrict__ xin_bf,  // sLSTM hs, bf16
                             const __bf16* __restrict__ Wpack,
                             const __bf16* __restrict__ Upack,
                             const __bf16* __restrict__ Vpack,   // KT=512, NT=8
                             const float* __restrict__ Wb,
                             const float* __restrict__ Ub,
                             const float* __restrict__ Vb,
                             float* __restrict__ out) {
  __shared__ __align__(32) float gates[BT * G4H];             // 32 KB
  __shared__ __align__(32) float xt_s[BT * HDIM];             // 8 KB
  __shared__ __align__(32) float h_s[BT * HDIM];              // 8 KB
  __shared__ __align__(32) float c_s[BT * HDIM];              // 8 KB
  __shared__ __align__(32) float m_s[BT * HDIM];              // 8 KB
  __shared__ __align__(1024) __bf16 vbuf[2][8][VCHUNK * 32 * 16]; // 64 KB

  const int tid  = threadIdx.x;
  const int lane = tid & 31;
  const int wave = tid >> 5;
  const int m    = lane & 15;
  const int hi   = (lane >> 4) & 1;
  const int b0   = blockIdx.x * BT;

  for (int e = tid; e < BT * HDIM; e += 256) { c_s[e] = 0.f; h_s[e] = 0.f; }
  __syncthreads();

  const int hcol = tid & (HDIM - 1);
  const float bi = Wb[hcol]            + Ub[hcol];
  const float bf = Wb[HDIM + hcol]     + Ub[HDIM + hcol];
  const float bo = Wb[2 * HDIM + hcol] + Ub[2 * HDIM + hcol];
  const float bg = Wb[3 * HDIM + hcol] + Ub[3 * HDIM + hcol];

  // this wave's 512 KB slice of packed Vw (n-tile = wave), kt-contiguous
  const char* vsrc = (const char*)Vpack + (size_t)wave * 512 * 1024;

  for (int t = 0; t < TLEN; ++t) {
    // stage x_t (sLSTM output) into LDS as fp32
    for (int e = tid; e < BT * HDIM; e += 256) {
      const size_t gidx = ((size_t)(b0 + (e >> 7)) * TLEN + t) * HDIM + (e & 127);
      xt_s[e] = (float)xin_bf[gidx];
      if (t + 1 < TLEN) __builtin_prefetch(xin_bf + gidx + HDIM, 0, 1);
    }
    __syncthreads();

    // kick off first V chunk DMA; it hides behind the gate GEMM below
    tdm_load_1d(lds_off(&vbuf[0][wave][0]), vsrc, VBYTES);

    // gate GEMM: gates = x_t@Ww + h@Uw
    bf16x16 xf[4], hf[4];
#pragma unroll
    for (int kt = 0; kt < 4; ++kt) {
      xf[kt] = make_afrag_f32(&xt_s[m * HDIM], kt, hi);
      hf[kt] = make_afrag_f32(&h_s[m * HDIM], kt, hi);
    }
#pragma unroll
    for (int nti = 0; nti < 4; ++nti) {
      const int nt = wave * 4 + nti;
      f32x8 acc = {};
#pragma unroll
      for (int kt = 0; kt < 4; ++kt) {
        acc = wmma_bf16(xf[kt], load_bfrag(Wpack, 4, kt, nt, lane), acc);
        acc = wmma_bf16(hf[kt], load_bfrag(Upack, 4, kt, nt, lane), acc);
      }
      const int n = (nt << 4) + m;
#pragma unroll
      for (int r = 0; r < 8; ++r)
        gates[(r + hi * 8) * G4H + n] = acc[r];
    }

    // bilinear GEMM with TDM double buffering (TENSORcnt is per-wave, in-order)
    {
      f32x8 macc = {};
      const float* xr = &xt_s[m * HDIM];
      const float* hr = &h_s[m * HDIM];
      for (int c = 0; c < VNCH - 1; ++c) {
        tdm_load_1d(lds_off(&vbuf[(c + 1) & 1][wave][0]),
                    vsrc + (size_t)(c + 1) * VBYTES, VBYTES);
        __builtin_amdgcn_s_wait_tensorcnt(1);   // chunk c landed (in-order)
        const __bf16* vb = &vbuf[c & 1][wave][0];
#pragma unroll
        for (int q = 0; q < VCHUNK; ++q) {
          const int kt = c * VCHUNK + q;
          const float xv = xr[kt >> 2];
          const int j0 = (kt & 3) * 32 + hi * 8;
          bf16x16 af;
#pragma unroll
          for (int j = 0; j < 8; ++j) {
            af[j]     = (__bf16)(xv * hr[j0 + j]);
            af[8 + j] = (__bf16)(xv * hr[j0 + 16 + j]);
          }
          macc = wmma_bf16(af, *(const bf16x16*)(vb + ((q * 32 + lane) << 4)), macc);
        }
      }
      // peeled last chunk
      __builtin_amdgcn_s_wait_tensorcnt(0);
      {
        const int c = VNCH - 1;
        const __bf16* vb = &vbuf[c & 1][wave][0];
#pragma unroll
        for (int q = 0; q < VCHUNK; ++q) {
          const int kt = c * VCHUNK + q;
          const float xv = xr[kt >> 2];
          const int j0 = (kt & 3) * 32 + hi * 8;
          bf16x16 af;
#pragma unroll
          for (int j = 0; j < 8; ++j) {
            af[j]     = (__bf16)(xv * hr[j0 + j]);
            af[8 + j] = (__bf16)(xv * hr[j0 + 16 + j]);
          }
          macc = wmma_bf16(af, *(const bf16x16*)(vb + ((q * 32 + lane) << 4)), macc);
        }
      }
      const int n = (wave << 4) + m;
      const float vbn = Vb[n];
#pragma unroll
      for (int r = 0; r < 8; ++r)
        m_s[(r + hi * 8) * HDIM + n] = tanhf(macc[r] + vbn);
    }
    __syncthreads();

    for (int e = tid; e < BT * HDIM; e += 256) {
      const int bb = e >> 7;
      const float* g = &gates[bb * G4H];
      const float iv = sigm(g[hcol] + bi);
      const float fv = sigm(g[HDIM + hcol] + bf);
      const float ov = sigm(g[2 * HDIM + hcol] + bo);
      const float gv = tanhf(g[3 * HDIM + hcol] + bg);
      const float cv = fv * c_s[e] + iv * gv + 0.1f * m_s[e];
      c_s[e] = cv;
      const float hv = ov * tanhf(cv);
      h_s[e] = hv;
      out[((size_t)(b0 + bb) * TLEN + t) * HDIM + hcol] = hv;
    }
    __syncthreads();
  }
}

// ---------------- host launch ----------------

extern "C" void kernel_launch(void* const* d_in, const int* in_sizes, int n_in,
                              void* d_out, int out_size, void* d_ws, size_t ws_size,
                              hipStream_t stream) {
  const float* x   = (const float*)d_in[0];
  const float* sWw = (const float*)d_in[1];
  const float* sWb = (const float*)d_in[2];
  const float* sUw = (const float*)d_in[3];
  const float* sUb = (const float*)d_in[4];
  const float* sAl = (const float*)d_in[5];
  const float* mWw = (const float*)d_in[6];
  const float* mWb = (const float*)d_in[7];
  const float* mUw = (const float*)d_in[8];
  const float* mUb = (const float*)d_in[9];
  const float* mVw = (const float*)d_in[10];
  const float* mVb = (const float*)d_in[11];

  char* ws = (char*)d_ws;
  size_t off = 0;
  auto alloc = [&](size_t bytes) -> void* {
    void* p = ws + off;
    off = (off + bytes + 1023) & ~(size_t)1023;
    return p;
  };
  const size_t nBTH = (size_t)BATCH * TLEN * HDIM;
  __bf16* x_bf  = (__bf16*)alloc(nBTH * 2);                       // 32 MB
  __bf16* hs_bf = (__bf16*)alloc(nBTH * 2);                       // 32 MB
  __bf16* sWp   = (__bf16*)alloc((size_t)HDIM * G4H * 2);         // 128 KB
  __bf16* sUp   = (__bf16*)alloc((size_t)HDIM * G4H * 2);
  __bf16* mWp   = (__bf16*)alloc((size_t)HDIM * G4H * 2);
  __bf16* mUp   = (__bf16*)alloc((size_t)HDIM * G4H * 2);
  __bf16* mVp   = (__bf16*)alloc((size_t)HDIM * HDIM * HDIM * 2); // 4 MB

  cvt_bf16_kernel<<<(int)((nBTH + 255) / 256), 256, 0, stream>>>(x, x_bf, nBTH);

  const int smallPackThreads = (HDIM / 32) * (G4H / 16) * 32;     // 4096
  pack_b_kernel<<<(smallPackThreads + 255) / 256, 256, 0, stream>>>(sWw, sWp, HDIM, G4H);
  pack_b_kernel<<<(smallPackThreads + 255) / 256, 256, 0, stream>>>(sUw, sUp, HDIM, G4H);
  pack_b_kernel<<<(smallPackThreads + 255) / 256, 256, 0, stream>>>(mWw, mWp, HDIM, G4H);
  pack_b_kernel<<<(smallPackThreads + 255) / 256, 256, 0, stream>>>(mUw, mUp, HDIM, G4H);
  const int vPackThreads = ((HDIM * HDIM) / 32) * (HDIM / 16) * 32; // 131072
  pack_b_kernel<<<(vPackThreads + 255) / 256, 256, 0, stream>>>(mVw, mVp, HDIM * HDIM, HDIM);

  slstm_kernel<<<BATCH / BT, 256, 0, stream>>>(x_bf, sWp, sUp, sWb, sUb, sAl, hs_bf);
  mlstm_kernel<<<BATCH / BT, 256, 0, stream>>>(hs_bf, mWp, mUp, mVp, mWb, mUb, mVb,
                                               (float*)d_out);
}
